// Attention_14963666059557
// MI455X (gfx1250) — compile-verified, matrix operand order checked by reading, not run on hardware
//
#include <hip/hip_runtime.h>
#include <math.h>

// Sizes fixed by the reference
#define B_   32
#define T_   2048
#define HID_ 1024
#define OUT_ 1024
#define DV_  1024
#define H2_  512   // OUT/2

typedef __attribute__((ext_vector_type(16))) _Float16 v16h;
typedef __attribute__((ext_vector_type(8)))  float    v8f;

// ---------------------------------------------------------------------------
// Kernel A: fold the two linear score layers:
//   w_eff[i] = sum_j W1[i, j] * W2[j]        (i < 1024, j < 512)
//   c        = sum_j b1[j]   * W2[j] + b2
// ---------------------------------------------------------------------------
__global__ __launch_bounds__(256) void prep_weff_kernel(
    const float* __restrict__ W1, const float* __restrict__ b1,
    const float* __restrict__ W2, const float* __restrict__ b2,
    float* __restrict__ w_eff, float* __restrict__ c_out)
{
    const int i = blockIdx.x * 256 + threadIdx.x;
    if (i < OUT_) {
        float s = 0.0f;
        for (int j = 0; j < H2_; ++j) s += W1[i * H2_ + j] * W2[j];
        w_eff[i] = s;
    }
    if (i == 0) {
        float s = b2[0];
        for (int j = 0; j < H2_; ++j) s += b1[j] * W2[j];
        c_out[0] = s;
    }
}

// ---------------------------------------------------------------------------
// Kernel B: query projection Q = hidden @ Wq + bq via v_wmma_f32_16x16x32_f16.
// hidden: [32, 1024] f32, Wq: [1024, 1024] f32 (K x N, row-major), Q: [32,1024].
// One wave computes one 16x16 D tile; 8 waves per block cover 8 N-tiles.
// Grid: (8, 2) -> 64 N-tiles x 2 M-tiles.
// ---------------------------------------------------------------------------
__global__ __launch_bounds__(256) void qproj_wmma_kernel(
    const float* __restrict__ hidden, const float* __restrict__ Wq,
    const float* __restrict__ bq, float* __restrict__ Q)
{
    const int lane  = threadIdx.x & 31;
    const int wave  = threadIdx.x >> 5;
    const int nTile = blockIdx.x * 8 + wave;   // 0..63
    const int mTile = blockIdx.y;              // 0..1
    const int half  = lane >> 4;               // 0: lanes 0-15, 1: lanes 16-31
    const int l15   = lane & 15;
    const int m     = mTile * 16 + l15;        // A row held by this lane
    const int n     = nTile * 16 + l15;        // B/D column held by this lane

    v8f acc = {};
    for (int kk = 0; kk < HID_; kk += 32) {
        // Per ISA 16-bit A layout (16x32, MxK):
        //   lanes 0-15 : elems 0-7 -> K 0-7,  elems 8-15 -> K 16-23
        //   lanes 16-31: elems 0-7 -> K 8-15, elems 8-15 -> K 24-31
        const int kb0 = kk + (half ? 8  : 0);
        const int kb1 = kk + (half ? 24 : 16);
        v16h a, b;
        #pragma unroll
        for (int e = 0; e < 8; ++e) {
            a[e]     = (_Float16)hidden[m * HID_ + kb0 + e];
            a[e + 8] = (_Float16)hidden[m * HID_ + kb1 + e];
            // B (32x16, KxN) symmetric layout over column n
            b[e]     = (_Float16)Wq[(kb0 + e) * OUT_ + n];
            b[e + 8] = (_Float16)Wq[(kb1 + e) * OUT_ + n];
        }
        // (neg_a, A, neg_b, B, c_mod, C, reuse_a, reuse_b)
        acc = __builtin_amdgcn_wmma_f32_16x16x32_f16(
            false, a, false, b, (short)0, acc, false, false);
    }
    // D layout: VGPR r -> row mTile*16 + r + 8*half, column n
    #pragma unroll
    for (int r = 0; r < 8; ++r) {
        const int row = mTile * 16 + r + half * 8;
        Q[row * OUT_ + n] = acc[r] + bq[n];
    }
}

// ---------------------------------------------------------------------------
// Kernel C: scores[b,t] = tanh(key[b,t,:] + Q[b,:]) . w_eff + c
// One wave per t (8 waves per block). Streams key (256 MB) with b128 loads.
// Grid: (T/8, B).
// ---------------------------------------------------------------------------
__global__ __launch_bounds__(256) void score_kernel(
    const float* __restrict__ key, const float* __restrict__ Q,
    const float* __restrict__ w_eff, const float* __restrict__ c_ptr,
    float* __restrict__ scores)
{
    const int lane = threadIdx.x & 31;
    const int wave = threadIdx.x >> 5;
    const int b    = blockIdx.y;
    const int t    = blockIdx.x * 8 + wave;

    const float4* kp = (const float4*)(key + ((size_t)b * T_ + t) * OUT_);
    const float4* qp = (const float4*)(Q + (size_t)b * OUT_);
    const float4* wp = (const float4*)w_eff;

    float acc = 0.0f;
    #pragma unroll
    for (int i = 0; i < OUT_ / (32 * 4); ++i) {   // 8 iterations
        const int idx = i * 32 + lane;            // float4 index, coalesced
        const float4 k4 = kp[idx];
        const float4 q4 = qp[idx];
        const float4 w4 = wp[idx];
        acc += tanhf(k4.x + q4.x) * w4.x;
        acc += tanhf(k4.y + q4.y) * w4.y;
        acc += tanhf(k4.z + q4.z) * w4.z;
        acc += tanhf(k4.w + q4.w) * w4.w;
    }
    // wave32 butterfly reduction
    #pragma unroll
    for (int off = 16; off > 0; off >>= 1)
        acc += __shfl_xor(acc, off, 32);
    if (lane == 0)
        scores[(size_t)b * T_ + t] = acc + c_ptr[0];
}

// ---------------------------------------------------------------------------
// Kernel D: masked softmax over T per batch row. One block per batch.
// Writes attn directly into d_out (second output).
// ---------------------------------------------------------------------------
__global__ __launch_bounds__(256) void softmax_kernel(
    const float* __restrict__ scores, const int* __restrict__ seq_lens,
    float* __restrict__ attn)
{
    __shared__ float red[256];
    const int b   = blockIdx.x;
    const int tid = threadIdx.x;
    const int sl  = seq_lens[b];

    float v[T_ / 256];
    float m = -INFINITY;
    #pragma unroll
    for (int j = 0; j < T_ / 256; ++j) {
        const int t = tid + j * 256;
        const float s = (t < sl) ? scores[(size_t)b * T_ + t] : -INFINITY;
        v[j] = s;
        m = fmaxf(m, s);
    }
    red[tid] = m;
    __syncthreads();
    for (int s2 = 128; s2 > 0; s2 >>= 1) {
        if (tid < s2) red[tid] = fmaxf(red[tid], red[tid + s2]);
        __syncthreads();
    }
    m = red[0];
    __syncthreads();

    float sum = 0.0f;
    #pragma unroll
    for (int j = 0; j < T_ / 256; ++j) {
        v[j] = __expf(v[j] - m);   // exp(-inf) == 0 handles masked slots
        sum += v[j];
    }
    red[tid] = sum;
    __syncthreads();
    for (int s2 = 128; s2 > 0; s2 >>= 1) {
        if (tid < s2) red[tid] += red[tid + s2];
        __syncthreads();
    }
    const float inv = 1.0f / red[0];
    #pragma unroll
    for (int j = 0; j < T_ / 256; ++j)
        attn[(size_t)b * T_ + tid + j * 256] = v[j] * inv;
}

// ---------------------------------------------------------------------------
// Kernel E: partial context sums. Streams value (256 MB).
// Grid: (DV/256, 8 t-chunks, B). Each block: 256 d-columns over 256 t rows,
// attention tile staged in LDS. Deterministic (no atomics).
// ---------------------------------------------------------------------------
__global__ __launch_bounds__(256) void context_partial_kernel(
    const float* __restrict__ value, const float* __restrict__ attn,
    float* __restrict__ partial)
{
    __shared__ float a_s[256];
    const int b  = blockIdx.z;
    const int d  = blockIdx.x * 256 + threadIdx.x;
    const int t0 = blockIdx.y * 256;

    a_s[threadIdx.x] = attn[(size_t)b * T_ + t0 + threadIdx.x];
    __syncthreads();

    float acc = 0.0f;
    const float* vp = value + ((size_t)b * T_ + t0) * DV_ + d;
    for (int j = 0; j < 256; ++j)
        acc += a_s[j] * vp[(size_t)j * DV_];

    partial[((size_t)(b * 8 + blockIdx.y)) * DV_ + d] = acc;
}

// Kernel F: reduce the 8 partials per (b, d) into context (first output).
__global__ __launch_bounds__(256) void context_reduce_kernel(
    const float* __restrict__ partial, float* __restrict__ ctx)
{
    const int i = blockIdx.x * 256 + threadIdx.x;   // 0 .. B*DV-1
    const int b = i >> 10;
    const int d = i & (DV_ - 1);
    float s = 0.0f;
    #pragma unroll
    for (int j = 0; j < 8; ++j)
        s += partial[((size_t)(b * 8 + j)) * DV_ + d];
    ctx[i] = s;
}

// ---------------------------------------------------------------------------
extern "C" void kernel_launch(void* const* d_in, const int* in_sizes, int n_in,
                              void* d_out, int out_size, void* d_ws, size_t ws_size,
                              hipStream_t stream)
{
    const float* hidden   = (const float*)d_in[0];
    const float* key      = (const float*)d_in[1];
    const float* value    = (const float*)d_in[2];
    const int*   seq_lens = (const int*)  d_in[3];
    const float* Wq       = (const float*)d_in[4];
    const float* bq       = (const float*)d_in[5];
    const float* W1       = (const float*)d_in[6];
    const float* b1       = (const float*)d_in[7];
    const float* W2       = (const float*)d_in[8];
    const float* b2       = (const float*)d_in[9];

    float* out  = (float*)d_out;
    float* ctx  = out;                 // [B, DV]   (first tuple output)
    float* attn = out + B_ * DV_;      // [B, T]    (second tuple output)

    // Workspace layout (floats)
    float* ws      = (float*)d_ws;
    float* w_eff   = ws;                         // 1024
    float* c_val   = ws + 1024;                  // 1
    float* Q       = ws + 2048;                  // 32768
    float* scores  = Q + B_ * OUT_;              // 65536
    float* partial = scores + B_ * T_;           // 32*8*1024 = 262144

    prep_weff_kernel      <<<dim3(4),          256, 0, stream>>>(W1, b1, W2, b2, w_eff, c_val);
    qproj_wmma_kernel     <<<dim3(8, 2),       256, 0, stream>>>(hidden, Wq, bq, Q);
    score_kernel          <<<dim3(T_ / 8, B_), 256, 0, stream>>>(key, Q, w_eff, c_val, scores);
    softmax_kernel        <<<dim3(B_),         256, 0, stream>>>(scores, seq_lens, attn);
    context_partial_kernel<<<dim3(DV_ / 256, 8, B_), 256, 0, stream>>>(value, attn, partial);
    context_reduce_kernel <<<dim3((B_ * DV_) / 256), 256, 0, stream>>>(partial, ctx);
}
